// FitModel_77867757076858
// MI455X (gfx1250) — matrix operand order repaired and checked by reading, not compiled, not run on hardware
//
#include <hip/hip_runtime.h>

// ---------------------------------------------------------------------------
// MI455X (gfx1250) implementation.
// Heavy math = two per-edge GEMMs routed through v_wmma_f32_16x16x32_f16.
// E=100000 edges processed as 3125 tiles of 32 edges (one wave32 per tile,
// M=32 so every B tile fetched from L2 feeds TWO wmma ops).
// ---------------------------------------------------------------------------

typedef __attribute__((ext_vector_type(16))) _Float16 v16h;
typedef __attribute__((ext_vector_type(8)))  _Float16 v8h;
typedef __attribute__((ext_vector_type(8)))  float    v8f;

#define NSD   48
#define NVD   10
#define NNODE 20000
#define NEDGE 100000
#define KDIM  144          // 3*NS
#define KPAD  160          // K padded to multiple of 32
#define WN    2784         // NS*NS + NS*NV
#define ODIM  78           // NS + 3*NV
#define NCH1  9            // 144/16 output chunks of GEMM1
#define NCH2  174          // 2784/16 output chunks of GEMM2
#define NKT   5            // 160/32 K-tiles
#define TILE_HALVES 512    // one 32x16 f16 B tile
#define MT    32           // edges per wave (two 16-row WMMA tiles)

// workspace layout (bytes)
#define WS_COUNTS 0                 // 20000 floats
#define WS_FC1T   81920             // 9*5*512 halves  = 46080 B
#define WS_FC2T   128000            // 174*5*512 halves = 890880 B
// total required ws: ~1.02 MB

// ---------------------------------------------------------------------------
// Kernel 0: zero output + counts
__global__ void zero_kernel(float* out, float* counts) {
    int i = blockIdx.x * blockDim.x + threadIdx.x;
    if (i < NNODE * ODIM) out[i] = 0.0f;
    if (i < NNODE) counts[i] = 0.0f;
}

// Kernel 1: per-source-node edge counts
__global__ void count_kernel(const int* edge_index, float* counts) {
    int e = blockIdx.x * blockDim.x + threadIdx.x;
    if (e < NEDGE) atomicAdd(&counts[edge_index[e]], 1.0f);
}

// ---------------------------------------------------------------------------
// Kernel 2: convert fc1_w / fc2_w (f32 row-major [K][N]) into f16 B-tiles
// laid out for the v_wmma_f32_16x16x32_f16 B operand in contiguous k order (as load_a):
//   tile (n-chunk c, k-tile t), element (lane, idx):
//     k = t*32 + (lane>=16 ? 16 : 0) + idx,  n = c*16 + (lane & 15)
//   stored at tileBase + lane*16 + idx   -> each lane loads 32 contiguous B.
__global__ void prep_weights_kernel(const float* fc1_w, const float* fc2_w,
                                    _Float16* fc1T, _Float16* fc2T) {
    int f = blockIdx.x * blockDim.x + threadIdx.x;
    const int total1 = NCH1 * NKT * TILE_HALVES;
    const int total2 = NCH2 * NKT * TILE_HALVES;
    if (f < total1) {
        int tile = f / TILE_HALVES, w = f % TILE_HALVES;
        int lane = w >> 4, idx = w & 15;
        int c = tile / NKT, t = tile % NKT;
        int k = t * 32 + ((lane >> 4) << 4) + idx;
        int n = c * 16 + (lane & 15);
        float v = (k < KDIM) ? fc1_w[k * KDIM + n] : 0.0f;
        fc1T[f] = (_Float16)v;
    } else if (f < total1 + total2) {
        int g = f - total1;
        int tile = g / TILE_HALVES, w = g % TILE_HALVES;
        int lane = w >> 4, idx = w & 15;
        int c = tile / NKT, t = tile % NKT;
        int k = t * 32 + ((lane >> 4) << 4) + idx;
        int n = c * 16 + (lane & 15);
        float v = (k < KDIM) ? fc2_w[k * WN + n] : 0.0f;
        fc2T[g] = (_Float16)v;
    }
}

// ---------------------------------------------------------------------------
// A-operand loader: row-major f16 matrix (stride KPAD halves) in LDS.
// Contiguous-k A layout used here (matches B tiles): lanes 0-15 hold M=lane, K = k0+0..15;
// lanes 16-31 hold M=lane-16, K = k0+16..31.
__device__ __forceinline__ v16h load_a(const _Float16* rowp, int k0, int koff) {
    v8h lo = *(const v8h*)(rowp + k0 + 2 * koff);
    v8h hi = *(const v8h*)(rowp + k0 + 2 * koff + 8);
    return __builtin_shufflevector(lo, hi, 0, 1, 2, 3, 4, 5, 6, 7,
                                           8, 9, 10, 11, 12, 13, 14, 15);
}

// ---------------------------------------------------------------------------
// Kernel 3: fused per-edge-tile pipeline. 1 wave32 == 32 edges (M=32).
__global__ __launch_bounds__(32)
void edge_tile_kernel(const float* __restrict__ node_attr,
                      const int*   __restrict__ edge_index,
                      const float* __restrict__ edge_attr,
                      const float* __restrict__ edge_sh,
                      const float* __restrict__ fc1_b,
                      const float* __restrict__ fc2_b,
                      const _Float16* __restrict__ fc1T,
                      const _Float16* __restrict__ fc2T,
                      float* __restrict__ out) {
    __shared__ _Float16 EAs[MT][KPAD];   // gathered input, f16
    __shared__ _Float16 Hs[MT][KPAD];    // relu(GEMM1), f16
    __shared__ float    xs[MT][NSD];     // node_attr[dst], f32
    __shared__ float    t1s[MT][NVD];    // x . w1 accumulators
    __shared__ float    sh0s[MT];
    __shared__ float    sh1s[MT][3];
    __shared__ int      srcs[MT];
    __shared__ int      dsts[MT];

    const int   lane = threadIdx.x;
    const int   e0   = blockIdx.x * MT;
    const float norm = 0.144337567297406441f; // 1/sqrt(48)

    srcs[lane] = edge_index[e0 + lane];
    dsts[lane] = edge_index[NEDGE + e0 + lane];
    sh0s[lane] = edge_sh[(e0 + lane) * 9];
    __syncthreads();

    for (int idx = lane; idx < MT * 3; idx += 32) {
        int r = idx / 3, m = idx % 3;
        sh1s[r][m] = edge_sh[(e0 + r) * 9 + 1 + m];
    }
    for (int idx = lane; idx < MT * NVD; idx += 32)
        t1s[idx / NVD][idx % NVD] = 0.0f;
    for (int idx = lane; idx < MT * NSD; idx += 32) {
        int r = idx / NSD, c = idx % NSD;
        xs[r][c] = node_attr[dsts[r] * NSD + c];
    }
    // gather ea = [edge_attr | node_attr[src] | node_attr[dst] | 0-pad]
    for (int idx = lane; idx < MT * KPAD; idx += 32) {
        int r = idx / KPAD, c = idx % KPAD;
        float v;
        if (c < NSD)            v = edge_attr[(e0 + r) * NSD + c];
        else if (c < 2 * NSD)   v = node_attr[srcs[r] * NSD + (c - NSD)];
        else if (c < 3 * NSD)   v = node_attr[dsts[r] * NSD + (c - 2 * NSD)];
        else                    v = 0.0f;
        EAs[r][c] = (_Float16)v;
    }
    // zero K-pad region of H once
    for (int idx = lane; idx < MT * (KPAD - KDIM); idx += 32)
        Hs[idx / (KPAD - KDIM)][KDIM + idx % (KPAD - KDIM)] = (_Float16)0.0f;
    __syncthreads();

    const int col     = lane & 15;   // N column of C/D tile; also A-row id
    const int half    = lane >> 4;
    const int rowbase = half * 8;    // C/D rows this lane owns (first M-tile)
    const int koff    = half * 8;    // A-operand K sub-offset

    // ---------------- GEMM1: H = relu(EA @ fc1 + b1) ----------------
    for (int c1 = 0; c1 < NCH1; ++c1) {
        v8f accA = {}, accB = {};
        const _Float16* bt = fc1T + (c1 * NKT) * TILE_HALVES;
#pragma unroll
        for (int t = 0; t < NKT; ++t) {
            v16h b  = *((const v16h*)(bt + t * TILE_HALVES) + lane);
            v16h aA = load_a(&EAs[col][0],      t * 32, koff);
            v16h aB = load_a(&EAs[16 + col][0], t * 32, koff);
            accA = __builtin_amdgcn_wmma_f32_16x16x32_f16(
                false, aA, false, b, (short)0, accA, false, false);
            accB = __builtin_amdgcn_wmma_f32_16x16x32_f16(
                false, aB, false, b, (short)0, accB, false, false);
        }
        float bias = fc1_b[c1 * 16 + col];
#pragma unroll
        for (int v = 0; v < 8; ++v) {
            float hA = fmaxf(accA[v] + bias, 0.0f);
            float hB = fmaxf(accB[v] + bias, 0.0f);
            Hs[rowbase + v][c1 * 16 + col]      = (_Float16)hA;
            Hs[16 + rowbase + v][c1 * 16 + col] = (_Float16)hB;
        }
    }
    __syncthreads();

    // -------- GEMM2 streamed + fused x-contraction (never build w) --------
    v8f a0A = {}, a1A = {}, a2A = {};
    v8f a0B = {}, a1B = {}, a2B = {};
    float xrA[8], xrB[8];
    int cur_i = -1;
    for (int c = 0; c < NCH2; ++c) {
        v8f TA = {}, TB = {};
        const _Float16* bt = fc2T + (c * NKT) * TILE_HALVES;
#pragma unroll
        for (int t = 0; t < NKT; ++t) {
            v16h b  = *((const v16h*)(bt + t * TILE_HALVES) + lane);
            v16h aA = load_a(&Hs[col][0],      t * 32, koff);
            v16h aB = load_a(&Hs[16 + col][0], t * 32, koff);
            TA = __builtin_amdgcn_wmma_f32_16x16x32_f16(
                false, aA, false, b, (short)0, TA, false, false);
            TB = __builtin_amdgcn_wmma_f32_16x16x32_f16(
                false, aB, false, b, (short)0, TB, false, false);
        }
        float bias = fc2_b[c * 16 + col];
        if (c < 144) {
            // w0 region: cols n=c*16+col -> i=c/3 (uniform), k=(c%3)*16+col
            int i = c / 3, r = c - 3 * i;
            if (i != cur_i) {
                cur_i = i;
#pragma unroll
                for (int v = 0; v < 8; ++v) {
                    xrA[v] = xs[rowbase + v][i];
                    xrB[v] = xs[16 + rowbase + v][i];
                }
            }
            if (r == 0) {
#pragma unroll
                for (int v = 0; v < 8; ++v) {
                    a0A[v] += xrA[v] * (TA[v] + bias);
                    a0B[v] += xrB[v] * (TB[v] + bias);
                }
            } else if (r == 1) {
#pragma unroll
                for (int v = 0; v < 8; ++v) {
                    a1A[v] += xrA[v] * (TA[v] + bias);
                    a1B[v] += xrB[v] * (TB[v] + bias);
                }
            } else {
#pragma unroll
                for (int v = 0; v < 8; ++v) {
                    a2A[v] += xrA[v] * (TA[v] + bias);
                    a2B[v] += xrB[v] * (TB[v] + bias);
                }
            }
        } else {
            // w1 region: nl = (c-144)*16+col -> i=nl/10, k=nl%10 (per-lane consts)
            int nl = (c - 144) * 16 + col;
            int i = nl / NVD, k = nl - NVD * i;
#pragma unroll
            for (int v = 0; v < 8; ++v) {
                atomicAdd(&t1s[rowbase + v][k],
                          xs[rowbase + v][i] * (TA[v] + bias));
                atomicAdd(&t1s[16 + rowbase + v][k],
                          xs[16 + rowbase + v][i] * (TB[v] + bias));
            }
        }
    }
    __syncthreads();

    // ---------------- epilogue: scale + scatter-add to out ----------------
#pragma unroll
    for (int v = 0; v < 8; ++v) {
        int rowA = rowbase + v, rowB = 16 + rowbase + v;
        float sA = sh0s[rowA] * norm;
        float sB = sh0s[rowB] * norm;
        int baseA = srcs[rowA] * ODIM;
        int baseB = srcs[rowB] * ODIM;
        atomicAdd(&out[baseA + col],      a0A[v] * sA);
        atomicAdd(&out[baseA + 16 + col], a1A[v] * sA);
        atomicAdd(&out[baseA + 32 + col], a2A[v] * sA);
        atomicAdd(&out[baseB + col],      a0B[v] * sB);
        atomicAdd(&out[baseB + 16 + col], a1B[v] * sB);
        atomicAdd(&out[baseB + 32 + col], a2B[v] * sB);
    }
    for (int idx = lane; idx < MT * NVD * 3; idx += 32) {
        int row = idx / (NVD * 3);
        int r = idx - row * (NVD * 3);
        int k = r / 3, m = r - 3 * k;
        float val = t1s[row][k] * sh1s[row][m] * norm;
        atomicAdd(&out[srcs[row] * ODIM + NSD + k * 3 + m], val);
    }
}

// Kernel 4: divide by per-node edge counts (segment mean)
__global__ void finalize_kernel(float* out, const float* counts) {
    int i = blockIdx.x * blockDim.x + threadIdx.x;
    if (i < NNODE * ODIM) out[i] /= fmaxf(counts[i / ODIM], 1.0f);
}

// ---------------------------------------------------------------------------
extern "C" void kernel_launch(void* const* d_in, const int* in_sizes, int n_in,
                              void* d_out, int out_size, void* d_ws, size_t ws_size,
                              hipStream_t stream) {
    const float* node_attr  = (const float*)d_in[0];
    const int*   edge_index = (const int*)  d_in[1];
    const float* edge_attr  = (const float*)d_in[2];
    const float* edge_sh    = (const float*)d_in[3];
    const float* fc1_w      = (const float*)d_in[4];
    const float* fc1_b      = (const float*)d_in[5];
    const float* fc2_w      = (const float*)d_in[6];
    const float* fc2_b      = (const float*)d_in[7];
    float* out = (float*)d_out;

    float*    counts = (float*)d_ws;
    _Float16* fc1T   = (_Float16*)((char*)d_ws + WS_FC1T);
    _Float16* fc2T   = (_Float16*)((char*)d_ws + WS_FC2T);

    const int zn = NNODE * ODIM;
    zero_kernel<<<(zn + 255) / 256, 256, 0, stream>>>(out, counts);
    count_kernel<<<(NEDGE + 255) / 256, 256, 0, stream>>>(edge_index, counts);

    const int prep_total = (NCH1 + NCH2) * NKT * TILE_HALVES;
    prep_weights_kernel<<<(prep_total + 255) / 256, 256, 0, stream>>>(
        fc1_w, fc2_w, fc1T, fc2T);

    edge_tile_kernel<<<NEDGE / MT, 32, 0, stream>>>(
        node_attr, edge_index, edge_attr, edge_sh, fc1_b, fc2_b,
        fc1T, fc2T, out);

    finalize_kernel<<<(zn + 255) / 256, 256, 0, stream>>>(out, counts);
}
